// MultiHeadAttention_3796751090171
// MI455X (gfx1250) — matrix operand order repaired and checked by reading, not compiled, not run on hardware
//
#include <hip/hip_runtime.h>
#include <hip/hip_bf16.h>

// ---------------------------------------------------------------------------
// MHA with per-head positional bias + cross-batch mask, flash-attention style.
// All GEMMs use v_wmma_f32_16x16x32_f16 (f32 accumulate).
//   N=3072, DIM=512, H=8, d=64, BATCH=8
// ---------------------------------------------------------------------------

#define NN     3072
#define DIM    512
#define NHEAD  8
#define HDIM   64
#define LDS_P  40   // padded row stride (halves): 80B rows -> 16B-aligned ds_load_b128

typedef __attribute__((ext_vector_type(16))) _Float16 v16h;
typedef __attribute__((ext_vector_type(8)))  _Float16 v8h;
typedef __attribute__((ext_vector_type(8)))  float    v8f;

__device__ __forceinline__ v8f wmma16(v16h a, v16h b, v8f c) {
  // (neg_a, A, neg_b, B, c_mod, C, reuse_a, reuse_b)
  return __builtin_amdgcn_wmma_f32_16x16x32_f16(false, a, false, b, (short)0, c,
                                                false, false);
}

// ----- A fragment (16x32 f16, M x K), f32 source row-major, ld mult of 32.
// lanes 0-15: row=lane,    halves 0..7 = K 0..7,  8..15 = K 16..23
// lanes16-31: row=lane-16, halves 0..7 = K 8..15, 8..15 = K 24..31
// Per lane: two 32B-aligned runs of 8 floats -> vector loads.
__device__ __forceinline__ v16h load_a_f32(const float* base, int ld) {
  const int lane = threadIdx.x & 31;
  const int row  = lane & 15;
  const int kb   = (lane >> 4) << 3; // 0 or 8 elements
  const float* p = base + (size_t)row * ld + kb;
  v8f lo = *(const v8f*)p;          // 32B aligned
  v8f hi = *(const v8f*)(p + 16);   // 32B aligned
  v16h a;
#pragma unroll
  for (int i = 0; i < 8; ++i) { a[i] = (_Float16)lo[i]; a[8 + i] = (_Float16)hi[i]; }
  return a;
}

// Same, f16 source; two 16B-aligned runs of 8 halves -> b128 loads.
__device__ __forceinline__ v16h load_a_f16(const _Float16* base, int ld) {
  const int lane = threadIdx.x & 31;
  const int row  = lane & 15;
  const int kb   = (lane >> 4) << 3;
  const _Float16* p = base + (size_t)row * ld + kb;
  v8h lo = *(const v8h*)p;
  v8h hi = *(const v8h*)(p + 16);
  return __builtin_shufflevector(lo, hi, 0, 1, 2, 3, 4, 5, 6, 7,
                                 8, 9, 10, 11, 12, 13, 14, 15);
}

// ----- B fragment (32x16 f16, K x N) where the SOURCE is the transpose:
// src row-major [N x K]; B(k,n) = src[n][k] -> two 16B-aligned runs of 8 halves per lane.
// lanes 0-15: col=lane, K 0..7,16..23 ; lanes 16-31: col=lane-16, K 8..15,24..31.
__device__ __forceinline__ v16h load_bt_f16(const _Float16* base, int ld) {
  const int lane = threadIdx.x & 31;
  const int col  = lane & 15;
  const int kb   = (lane >> 4) << 3;
  const _Float16* p = base + (size_t)col * ld + kb;
  v8h lo = *(const v8h*)p;
  v8h hi = *(const v8h*)(p + 16);
  return __builtin_shufflevector(lo, hi, 0, 1, 2, 3, 4, 5, 6, 7,
                                 8, 9, 10, 11, 12, 13, 14, 15);
}

// ===========================================================================
// Kernel 0: weight prep — W(f32,[in][out]) -> Wt(f16,[out][in]) for all 4
// weight matrices (makes GEMM B-fragments contiguous 32B-per-lane loads).
// ===========================================================================
__global__ void mha_wt_kernel(const float* __restrict__ W0, const float* __restrict__ W1,
                              const float* __restrict__ W2, const float* __restrict__ W3,
                              _Float16* __restrict__ T0, _Float16* __restrict__ T1,
                              _Float16* __restrict__ T2, _Float16* __restrict__ T3) {
  const float* W;
  _Float16* T;
  switch (blockIdx.y) {
    case 0: W = W0; T = T0; break;
    case 1: W = W1; T = T1; break;
    case 2: W = W2; T = T2; break;
    default: W = W3; T = T3; break;
  }
  const int idx = blockIdx.x * blockDim.x + threadIdx.x; // 512*512
  const int n = idx >> 9, k = idx & 511;
  T[idx] = (_Float16)W[(size_t)k * DIM + n];             // Wt[n][k] = W[k][n]
}

// ===========================================================================
// Kernel 1: fused QKV projection  out = X(3072x512) * W + b
//   gridDim.y selects: 0 -> Q (scaled 1/8, [H][N][64])
//                      1 -> K ([H][N][64])
//                      2 -> V (transposed, [H][64][N])
// One wave = 16 rows x 64 cols.
// ===========================================================================
__global__ void mha_proj_kernel(const float* __restrict__ Xq, const float* __restrict__ Xk,
                                const float* __restrict__ Xv,
                                const _Float16* __restrict__ WtQ,
                                const _Float16* __restrict__ WtK,
                                const _Float16* __restrict__ WtV,
                                const float* __restrict__ bq, const float* __restrict__ bk,
                                const float* __restrict__ bvv,
                                _Float16* __restrict__ Qo, _Float16* __restrict__ Ko,
                                _Float16* __restrict__ Vo) {
  const int mode = blockIdx.y;
  const float* X;
  const _Float16* Wt;
  const float* bias;
  _Float16* out;
  switch (mode) {
    case 0: X = Xq; Wt = WtQ; bias = bq;  out = Qo; break;
    case 1: X = Xk; Wt = WtK; bias = bk;  out = Ko; break;
    default: X = Xv; Wt = WtV; bias = bvv; out = Vo; break;
  }

  const int w     = (blockIdx.x * blockDim.x + threadIdx.x) >> 5;
  const int lane  = threadIdx.x & 31;
  const int mbase = (w >> 3) * 16;
  const int nb    = (w & 7) * 64;

  v8f acc[4];
#pragma unroll
  for (int cb = 0; cb < 4; ++cb) acc[cb] = (v8f){0, 0, 0, 0, 0, 0, 0, 0};

  for (int kk = 0; kk < DIM; kk += 32) {
    v16h a = load_a_f32(X + (size_t)mbase * DIM + kk, DIM);
#pragma unroll
    for (int cb = 0; cb < 4; ++cb) {
      v16h b = load_bt_f16(Wt + (size_t)(nb + cb * 16) * DIM + kk, DIM);
      acc[cb] = wmma16(a, b, acc[cb]);
    }
  }

  const int g = lane >> 4, col = lane & 15;
#pragma unroll
  for (int cb = 0; cb < 4; ++cb) {
    const int c = nb + cb * 16 + col;
    const float bv = bias[c];
    const int h = c >> 6, d = c & 63;
#pragma unroll
    for (int v = 0; v < 8; ++v) {
      const int r = mbase + v + 8 * g;
      float val = acc[cb][v] + bv;
      if (mode == 0) val *= 0.125f; // fold 1/sqrt(HDIM) into Q
      size_t addr = (mode == 2)
                        ? ((size_t)h * HDIM * NN + (size_t)d * NN + r)
                        : ((size_t)h * NN * HDIM + (size_t)r * HDIM + d);
      out[addr] = (_Float16)val;
    }
  }
}

// ===========================================================================
// Kernel 2: fused flash attention.
// One wave = (head, 16-query tile). Streams pos_enc exactly once (302 MB).
// ===========================================================================
__global__ void mha_attn_kernel(const _Float16* __restrict__ Qb,
                                const _Float16* __restrict__ Kb,
                                const _Float16* __restrict__ Vt,
                                const float* __restrict__ pos,
                                const int* __restrict__ q_batch,
                                const int* __restrict__ k_batch,
                                _Float16* __restrict__ Hb) {
  __shared__ _Float16 lds[8][16 * LDS_P]; // per-wave P bounce tile (C->A layout)

  const int wl    = threadIdx.x >> 5;
  const int w     = (blockIdx.x * blockDim.x + threadIdx.x) >> 5;
  const int lane  = threadIdx.x & 31;
  const int h     = w / (NN / 16);
  const int qbase = (w % (NN / 16)) * 16;
  const int g = lane >> 4, col = lane & 15;

  const _Float16* Qh = Qb + (size_t)h * NN * HDIM + (size_t)qbase * HDIM;
  const _Float16* Kh = Kb + (size_t)h * NN * HDIM;
  const _Float16* Vh = Vt + (size_t)h * HDIM * NN;
  const float* posh  = pos + (size_t)h * NN * NN + (size_t)qbase * NN;

  // Q fragments for the full head dim (already scaled by 1/8)
  const v16h q0 = load_a_f16(Qh, HDIM);       // dims 0..31
  const v16h q1 = load_a_f16(Qh + 32, HDIM);  // dims 32..63

  int qb[8];
#pragma unroll
  for (int v = 0; v < 8; ++v) qb[v] = q_batch[qbase + v + 8 * g];

  float m[8], l[8];
  v8f O[4];
#pragma unroll
  for (int v = 0; v < 8; ++v) { m[v] = -3.0e38f; l[v] = 0.0f; }
#pragma unroll
  for (int cb = 0; cb < 4; ++cb) O[cb] = (v8f){0, 0, 0, 0, 0, 0, 0, 0};

  _Float16* P = &lds[wl][0];

  for (int kt = 0; kt < NN; kt += 32) {
    // Prefetch the pos tile 16 iterations ahead (dominant HBM stream).
    {
      const int pf = kt + 512;
      if (pf < NN)
        __builtin_prefetch(posh + (size_t)(lane & 15) * NN + pf + ((lane >> 4) << 4), 0, 1);
    }

    // ---- S = Q K^T for 32 keys (two 16-key column tiles) ----
    v8f s0 = (v8f){0, 0, 0, 0, 0, 0, 0, 0};
    v8f s1 = (v8f){0, 0, 0, 0, 0, 0, 0, 0};
    s0 = wmma16(q0, load_bt_f16(Kh + (size_t)kt * HDIM, HDIM), s0);
    s0 = wmma16(q1, load_bt_f16(Kh + (size_t)kt * HDIM + 32, HDIM), s0);
    s1 = wmma16(q0, load_bt_f16(Kh + (size_t)(kt + 16) * HDIM, HDIM), s1);
    s1 = wmma16(q1, load_bt_f16(Kh + (size_t)(kt + 16) * HDIM + 32, HDIM), s1);

    const int kb0 = k_batch[kt + col];
    const int kb1 = k_batch[kt + 16 + col];

    // ---- add pos bias (fp32 stream), apply cross-batch mask ----
    float t0[8], t1[8];
#pragma unroll
    for (int v = 0; v < 8; ++v) {
      const int r = v + 8 * g;
      const float p0 = posh[(size_t)r * NN + kt + col];
      const float p1 = posh[(size_t)r * NN + kt + 16 + col];
      t0[v] = (qb[v] == kb0) ? (s0[v] + p0) : -1.0e9f;
      t1[v] = (qb[v] == kb1) ? (s1[v] + p1) : -1.0e9f;
    }

    // ---- online softmax update (row reductions across 16-lane groups) ----
#pragma unroll
    for (int v = 0; v < 8; ++v) {
      float x = fmaxf(t0[v], t1[v]);
#pragma unroll
      for (int dlt = 1; dlt < 16; dlt <<= 1) x = fmaxf(x, __shfl_xor(x, dlt, 32));
      const float mn = fmaxf(m[v], x);
      const float alpha = __expf(m[v] - mn);
      m[v] = mn;
      t0[v] = __expf(t0[v] - mn);
      t1[v] = __expf(t1[v] - mn);
      float s = t0[v] + t1[v];
#pragma unroll
      for (int dlt = 1; dlt < 16; dlt <<= 1) s += __shfl_xor(s, dlt, 32);
      l[v] = l[v] * alpha + s;
#pragma unroll
      for (int cb = 0; cb < 4; ++cb) O[cb][v] *= alpha;
    }

    // ---- P: C-fragment layout -> A-fragment layout via per-wave LDS ----
#pragma unroll
    for (int v = 0; v < 8; ++v) {
      const int r = v + 8 * g;
      P[r * LDS_P + col]      = (_Float16)t0[v];
      P[r * LDS_P + 16 + col] = (_Float16)t1[v];
    }
    v16h pa;
    {
      const int row = lane & 15;
      const int kb  = (lane >> 4) << 3;
      const _Float16* pp = P + row * LDS_P + kb;   // 16B aligned (LDS_P=40)
      v8h lo = *(const v8h*)pp;
      v8h hi = *(const v8h*)(pp + 16);
      pa = __builtin_shufflevector(lo, hi, 0, 1, 2, 3, 4, 5, 6, 7,
                                   8, 9, 10, 11, 12, 13, 14, 15);
    }

    // ---- O += P (16x32) * V (32x64), V stored transposed [64][N] ----
#pragma unroll
    for (int cb = 0; cb < 4; ++cb) {
      v16h vb = load_bt_f16(Vh + (size_t)(cb * 16) * NN + kt, NN);
      O[cb] = wmma16(pa, vb, O[cb]);
    }
  }

  // ---- epilogue: H[n][h*64+d] = O / l  (f16 for the final GEMM) ----
#pragma unroll
  for (int cb = 0; cb < 4; ++cb) {
#pragma unroll
    for (int v = 0; v < 8; ++v) {
      const int r = qbase + v + 8 * g;
      Hb[(size_t)r * DIM + h * HDIM + cb * 16 + col] =
          (_Float16)(O[cb][v] / l[v]);
    }
  }
}

// ===========================================================================
// Kernel 3: output projection  out = H(3072x512,f16) * Wo + bo  (fp32 out)
// ===========================================================================
__global__ void mha_outproj_kernel(const _Float16* __restrict__ Hb,
                                   const _Float16* __restrict__ WtO,
                                   const float* __restrict__ bias,
                                   float* __restrict__ out) {
  const int w     = (blockIdx.x * blockDim.x + threadIdx.x) >> 5;
  const int lane  = threadIdx.x & 31;
  const int mbase = (w >> 3) * 16;
  const int nb    = (w & 7) * 64;

  v8f acc[4];
#pragma unroll
  for (int cb = 0; cb < 4; ++cb) acc[cb] = (v8f){0, 0, 0, 0, 0, 0, 0, 0};

  for (int kk = 0; kk < DIM; kk += 32) {
    v16h a = load_a_f16(Hb + (size_t)mbase * DIM + kk, DIM);
#pragma unroll
    for (int cb = 0; cb < 4; ++cb) {
      v16h b = load_bt_f16(WtO + (size_t)(nb + cb * 16) * DIM + kk, DIM);
      acc[cb] = wmma16(a, b, acc[cb]);
    }
  }

  const int g = lane >> 4, col = lane & 15;
#pragma unroll
  for (int cb = 0; cb < 4; ++cb) {
    const int c = nb + cb * 16 + col;
    const float bv = bias[c];
#pragma unroll
    for (int v = 0; v < 8; ++v) {
      const int r = mbase + v + 8 * g;
      out[(size_t)r * DIM + c] = acc[cb][v] + bv;
    }
  }
}

// ===========================================================================
extern "C" void kernel_launch(void* const* d_in, const int* in_sizes, int n_in,
                              void* d_out, int out_size, void* d_ws, size_t ws_size,
                              hipStream_t stream) {
  const float* q_feat  = (const float*)d_in[0];
  const float* k_feat  = (const float*)d_in[1];
  const float* v_feat  = (const float*)d_in[2];
  const float* pos_enc = (const float*)d_in[3];
  const int*   q_batch = (const int*)d_in[4];
  const int*   k_batch = (const int*)d_in[5];
  const float* Wq = (const float*)d_in[6];
  const float* bq = (const float*)d_in[7];
  const float* Wk = (const float*)d_in[8];
  const float* bk = (const float*)d_in[9];
  const float* Wv = (const float*)d_in[10];
  const float* bv = (const float*)d_in[11];
  const float* Wo = (const float*)d_in[12];
  const float* bo = (const float*)d_in[13];

  _Float16* Qb  = (_Float16*)d_ws;                // [H][N][64] f16 (pre-scaled)
  _Float16* Kb  = Qb + (size_t)NN * DIM;          // [H][N][64] f16
  _Float16* Vt  = Kb + (size_t)NN * DIM;          // [H][64][N] f16
  _Float16* Hb  = Vt + (size_t)NN * DIM;          // [N][512]   f16
  _Float16* WtQ = Hb + (size_t)NN * DIM;          // [out][in] f16 transposed
  _Float16* WtK = WtQ + (size_t)DIM * DIM;
  _Float16* WtV = WtK + (size_t)DIM * DIM;
  _Float16* WtO = WtV + (size_t)DIM * DIM;

  const dim3 blk(256);

  mha_wt_kernel<<<dim3(DIM * DIM / 256, 4), blk, 0, stream>>>(
      Wq, Wk, Wv, Wo, WtQ, WtK, WtV, WtO);

  mha_proj_kernel<<<dim3((NN / 16) * (DIM / 64) * 32 / 256, 3), blk, 0, stream>>>(
      q_feat, k_feat, v_feat, WtQ, WtK, WtV, bq, bk, bv, Qb, Kb, Vt);

  mha_attn_kernel<<<NHEAD * (NN / 16) * 32 / 256, blk, 0, stream>>>(
      Qb, Kb, Vt, pos_enc, q_batch, k_batch, Hb);

  mha_outproj_kernel<<<(NN / 16) * (DIM / 64) * 32 / 256, blk, 0, stream>>>(
      Hb, WtO, bo, (float*)d_out);
}
